// CapsNet_66202625901010
// MI455X (gfx1250) — compile-verified, matrix operand order checked by reading, not run on hardware
//
#include <hip/hip_runtime.h>
#include <cmath>

typedef __attribute__((ext_vector_type(16))) __bf16 v16bf;
typedef __attribute__((ext_vector_type(8)))  __bf16 v8bf;
typedef __attribute__((ext_vector_type(8)))  float  v8f;

#define ASTRIDE 40   // LDS row stride in bf16 elements (padded: 32 data + 8 pad)

// ---------------------------------------------------------------------------
// Weight fp32 -> bf16 (with K padding to Kpad)
// ---------------------------------------------------------------------------
__global__ void cvt_weight_k(const float* __restrict__ w, __bf16* __restrict__ wb,
                             int N, int Kdepth, int Kpad) {
    int idx = blockIdx.x * 256 + threadIdx.x;
    if (idx >= N * Kpad) return;
    int n = idx / Kpad, k = idx - n * Kpad;
    wb[idx] = (k < Kdepth) ? (__bf16)w[(size_t)n * Kdepth + k] : (__bf16)0.0f;
}

// ---------------------------------------------------------------------------
// Implicit-GEMM convolution via WMMA bf16 (9x9 kernels, Cout == 256).
// Block: 256 threads = 8 waves. Block tile: M=64 pixels x N=256 channels.
// Wave tile: 32 (M, two 16-row subtiles) x 64 (N, four 16-col subtiles)
//          = 8 x wmma_f32_16x16x32_bf16 accumulators; each B fragment is
//            reused by both M subtiles (8 WMMA per 8 B-loads per chunk).
// ---------------------------------------------------------------------------
template<typename InT, typename OutT, bool RELU, int KH, int KW>
__global__ __launch_bounds__(256)
void conv_wmma_k(const InT* __restrict__ in, const __bf16* __restrict__ wB,
                 const float* __restrict__ bias, OutT* __restrict__ out,
                 int Bn, int Cin, int Hin, int Win,
                 int Cout, int Hout, int Wout, int stride,
                 int Kdepth, int Kpad)
{
    __shared__ __align__(16) __bf16 Asm[64 * ASTRIDE];

    const int tid   = threadIdx.x;
    const int lane  = tid & 31;
    const int wave  = tid >> 5;
    const int waveM = wave >> 2;        // 0..1  (which 32-row half of the tile)
    const int waveN = wave & 3;         // 0..3  (which 64-col quarter)
    const int Mtot  = Bn * Hout * Wout;
    const int blockM = blockIdx.x * 64;

    // staging coordinates: each thread stages 8 consecutive k's for one pixel
    const int sm   = tid >> 2;          // 0..63  (M row inside tile)
    const int skk  = (tid & 3) * 8;     // 0,8,16,24 (k offset inside chunk)
    const int sgm  = blockM + sm;
    const bool svalid = (sgm < Mtot);
    int sp0 = 0;
    if (svalid) {
        int b   = sgm / (Hout * Wout);
        int rem = sgm - b * Hout * Wout;
        int oy  = rem / Wout;
        int ox  = rem - oy * Wout;
        sp0 = ((b * Cin) * Hin + oy * stride) * Win + ox * stride;
    }
    const int HW = Hin * Win;

    v8f acc[2][4] = {};
    const int lcol = lane & 15;
    const int lhal = lane >> 4;
    const int chunks = Kpad >> 5;

    for (int kc = 0; kc < chunks; ++kc) {
        __syncthreads();
        // ---- stage im2col A tile (64 pixels x 32 k) into LDS as bf16 ----
        #pragma unroll
        for (int j = 0; j < 8; ++j) {
            int k = (kc << 5) + skk + j;
            float vv = 0.0f;
            if (svalid && k < Kdepth) {
                int ci  = k / (KH * KW);
                int rem = k - ci * (KH * KW);
                int ky  = rem / KW;
                int kx  = rem - ky * KW;
                vv = (float)in[sp0 + ci * HW + ky * Win + kx];
            }
            Asm[sm * ASTRIDE + skk + j] = (__bf16)vv;
        }
        __syncthreads();

        // ---- two A fragments (16x32 bf16 each): lane<16 row=lcol K{0..7,16..23};
        //      lane>=16 row=lcol K{8..15,24..31} ----
        v16bf afrag[2];
        #pragma unroll
        for (int sub = 0; sub < 2; ++sub) {
            const __bf16* ap =
                &Asm[(waveM * 32 + sub * 16 + lcol) * ASTRIDE + lhal * 8];
            v8bf alo = *reinterpret_cast<const v8bf*>(ap);
            v8bf ahi = *reinterpret_cast<const v8bf*>(ap + 16);
            #pragma unroll
            for (int i = 0; i < 8; ++i) {
                afrag[sub][i] = alo[i];
                afrag[sub][i + 8] = ahi[i];
            }
        }

        #pragma unroll
        for (int s = 0; s < 4; ++s) {
            // ---- B fragment (32x16 bf16), column n, same K striping ----
            int n = waveN * 64 + s * 16 + lcol;
            const __bf16* wp = wB + (size_t)n * Kpad + (kc << 5) + lhal * 8;
            v8bf blo = *reinterpret_cast<const v8bf*>(wp);
            v8bf bhi = *reinterpret_cast<const v8bf*>(wp + 16);
            v16bf bfrag;
            #pragma unroll
            for (int i = 0; i < 8; ++i) { bfrag[i] = blo[i]; bfrag[i + 8] = bhi[i]; }

            acc[0][s] = __builtin_amdgcn_wmma_f32_16x16x32_bf16(
                false, afrag[0], false, bfrag, (short)0, acc[0][s], false, false);
            acc[1][s] = __builtin_amdgcn_wmma_f32_16x16x32_bf16(
                false, afrag[1], false, bfrag, (short)0, acc[1][s], false, false);
        }
    }

    // ---- epilogue: C layout lane<16 -> (m=vi, n=lcol); lane>=16 -> (m=vi+8) ----
    const int mo = lhal * 8;
    #pragma unroll
    for (int sub = 0; sub < 2; ++sub) {
        #pragma unroll
        for (int s = 0; s < 4; ++s) {
            int n = waveN * 64 + s * 16 + lcol;
            float bb = bias[n];
            #pragma unroll
            for (int vi = 0; vi < 8; ++vi) {
                int gm = blockM + waveM * 32 + sub * 16 + mo + vi;
                if (gm < Mtot) {
                    float val = acc[sub][s][vi] + bb;
                    if (RELU) val = fmaxf(val, 0.0f);
                    int b   = gm / (Hout * Wout);
                    int rem = gm - b * Hout * Wout;
                    int oy  = rem / Wout;
                    int ox  = rem - oy * Wout;
                    out[((size_t)(b * Cout + n) * Hout + oy) * Wout + ox] = (OutT)val;
                }
            }
        }
    }
}

// ---------------------------------------------------------------------------
// Squash (in-place) over groups of 32 contiguous floats: 8*12800 capsules
// ---------------------------------------------------------------------------
__global__ void squash_k(float* __restrict__ u) {
    int idx = blockIdx.x * 256 + threadIdx.x;
    if (idx >= 102400) return;
    float4* p = reinterpret_cast<float4*>(u + (size_t)idx * 32);
    float4 t[8];
    float sn = 0.0f;
    #pragma unroll
    for (int i = 0; i < 8; ++i) {
        t[i] = p[i];
        sn += t[i].x * t[i].x + t[i].y * t[i].y + t[i].z * t[i].z + t[i].w * t[i].w;
    }
    float sc = (sn > 0.0f) ? sn / ((1.0f + sn) * sqrtf(sn)) : 0.0f;
    #pragma unroll
    for (int i = 0; i < 8; ++i) {
        t[i].x *= sc; t[i].y *= sc; t[i].z *= sc; t[i].w *= sc;
        p[i] = t[i];
    }
}

// ---------------------------------------------------------------------------
// u_hat[b,r,c] = dot32(digit_W[r,c,0,:], u[b,r,:])
// ---------------------------------------------------------------------------
__global__ void uhat_k(const float* __restrict__ dW, const float* __restrict__ u,
                       float* __restrict__ uh) {
    int idx = blockIdx.x * 256 + threadIdx.x;
    if (idx >= 1024000) return;
    int b   = idx / 128000;
    int rem = idx - b * 128000;
    int r   = rem / 10;
    int c   = rem - r * 10;
    const float* Wp = dW + ((size_t)r * 10 + c) * 32;
    const float* Up = u + (size_t)b * 409600 + (size_t)r * 32;
    float a = 0.0f;
    #pragma unroll
    for (int i = 0; i < 32; ++i) a += Wp[i] * Up[i];
    uh[idx] = a;  // layout: (b*12800 + r)*10 + c == idx
}

__global__ void zero_k(float* __restrict__ p, int n) {
    int idx = blockIdx.x * 256 + threadIdx.x;
    if (idx < n) p[idx] = 0.0f;
}

// ---------------------------------------------------------------------------
// Routing step A: per class c — softmax over r of b_ij, weighted sum s_j,
// scalar squash -> v[b,c]. One block per class.
// ---------------------------------------------------------------------------
__global__ void routing1_k(const float* __restrict__ bij, const float* __restrict__ uh,
                           float* __restrict__ v, float* __restrict__ vout) {
    const int c = blockIdx.x;     // 0..9
    const int tid = threadIdx.x;  // 256
    __shared__ float sred[256];
    __shared__ float s9[9 * 256];

    float mx = -3.0e38f;
    for (int r = tid; r < 12800; r += 256) mx = fmaxf(mx, bij[r * 10 + c]);
    sred[tid] = mx;
    __syncthreads();
    for (int s = 128; s > 0; s >>= 1) {
        if (tid < s) sred[tid] = fmaxf(sred[tid], sred[tid + s]);
        __syncthreads();
    }
    const float M = sred[0];
    __syncthreads();

    float z = 0.0f, S[8] = {0, 0, 0, 0, 0, 0, 0, 0};
    for (int r = tid; r < 12800; r += 256) {
        float e = __expf(bij[r * 10 + c] - M);
        z += e;
        #pragma unroll
        for (int b = 0; b < 8; ++b) S[b] += e * uh[(size_t)b * 128000 + r * 10 + c];
    }
    s9[tid] = z;
    #pragma unroll
    for (int b = 0; b < 8; ++b) s9[(b + 1) * 256 + tid] = S[b];
    __syncthreads();
    if (tid < 9) {
        float a = 0.0f;
        for (int i = 0; i < 256; ++i) a += s9[tid * 256 + i];
        s9[tid * 256] = a;
    }
    __syncthreads();
    if (tid == 0) {
        float Z = s9[0];
        for (int b = 0; b < 8; ++b) {
            float s = s9[(b + 1) * 256] / Z;
            float val = s * fabsf(s) / (1.0f + s * s);   // squash of a scalar
            v[b * 10 + c] = val;
            vout[b * 10 + c] = val;
        }
    }
}

// ---------------------------------------------------------------------------
// Routing step B: b_ij[r,c] += sum_b u_hat[b,r,c] * v[b,c]
// ---------------------------------------------------------------------------
__global__ void routing2_k(const float* __restrict__ uh, const float* __restrict__ v,
                           float* __restrict__ bij) {
    int idx = blockIdx.x * 256 + threadIdx.x;
    if (idx >= 128000) return;
    int r = idx / 10, c = idx - r * 10;
    float a = 0.0f;
    #pragma unroll
    for (int b = 0; b < 8; ++b)
        a += uh[(size_t)b * 128000 + r * 10 + c] * v[b * 10 + c];
    bij[idx] += a;
}

// ---------------------------------------------------------------------------
// Decoder
// ---------------------------------------------------------------------------
__global__ void dec1_k(const float* __restrict__ v, const float* __restrict__ w1,
                       const float* __restrict__ b1, float* __restrict__ r1) {
    int idx = blockIdx.x * 256 + threadIdx.x;      // 8*512
    if (idx >= 4096) return;
    int b = idx >> 9, j = idx & 511;
    float a = b1[j];
    #pragma unroll
    for (int i = 0; i < 10; ++i) a += v[b * 10 + i] * w1[i * 512 + j];
    r1[idx] = fmaxf(a, 0.0f);
}

__global__ void dec2_k(const float* __restrict__ r1, const float* __restrict__ w2,
                       const float* __restrict__ b2, float* __restrict__ r2) {
    int idx = blockIdx.x * 256 + threadIdx.x;      // 8*1024
    if (idx >= 8192) return;
    int b = idx >> 10, j = idx & 1023;
    float a = b2[j];
    for (int i = 0; i < 512; ++i) a += r1[b * 512 + i] * w2[i * 1024 + j];
    r2[idx] = fmaxf(a, 0.0f);
}

__global__ __launch_bounds__(256)
void dec3_k(const float* __restrict__ r2, const float* __restrict__ w3,
            const float* __restrict__ b3, float* __restrict__ outp) {
    __shared__ float s[8192];                      // all 8x1024 activations
    const int tid = threadIdx.x;
    const int n = blockIdx.x * 256 + tid;          // 108*256 == 27648 exactly
    for (int i = tid; i < 8192; i += 256) s[i] = r2[i];
    __syncthreads();
    float acc[8] = {0, 0, 0, 0, 0, 0, 0, 0};
    for (int k = 0; k < 1024; ++k) {
        float w = w3[(size_t)k * 27648 + n];       // coalesced across threads
        #pragma unroll
        for (int b = 0; b < 8; ++b) acc[b] += s[b * 1024 + k] * w;
    }
    float bb = b3[n];
    #pragma unroll
    for (int b = 0; b < 8; ++b)
        outp[(size_t)b * 27648 + n] = 1.0f / (1.0f + __expf(-(acc[b] + bb)));
}

// ---------------------------------------------------------------------------
// Host launcher
// ---------------------------------------------------------------------------
extern "C" void kernel_launch(void* const* d_in, const int* in_sizes, int n_in,
                              void* d_out, int out_size, void* d_ws, size_t ws_size,
                              hipStream_t stream) {
    const float* x   = (const float*)d_in[0];
    const float* c1w = (const float*)d_in[1];
    const float* c1b = (const float*)d_in[2];
    const float* pw  = (const float*)d_in[3];
    const float* pb  = (const float*)d_in[4];
    const float* dW  = (const float*)d_in[5];
    const float* w1  = (const float*)d_in[6];
    const float* b1  = (const float*)d_in[7];
    const float* w2  = (const float*)d_in[8];
    const float* b2  = (const float*)d_in[9];
    const float* w3  = (const float*)d_in[10];
    const float* b3  = (const float*)d_in[11];
    float* out = (float*)d_out;

    char* ws = (char*)d_ws;
    __bf16* w1b = (__bf16*)(ws + 0);           //   131072 B  (256 x 256)
    __bf16* w2b = (__bf16*)(ws + 131072);      // 10616832 B  (256 x 20736)
    __bf16* h   = (__bf16*)(ws + 10747904);    // 31719424 B  (8,256,88,88)
    float*  u   = (float*)(ws + 42467328);     // 13107200 B  (8,256,40,40)
    float*  uh  = (float*)(ws + 55574528);     //  4096000 B  (8,12800,10)
    float*  bij = (float*)(ws + 59670528);     //   512000 B  (12800,10)
    float*  v   = (float*)(ws + 60182528);     //      512 B  (8,10)
    float*  r1  = (float*)(ws + 60183040);     //    16384 B  (8,512)
    float*  r2  = (float*)(ws + 60199424);     //    32768 B  (8,1024)

    // bf16 weight conversion (padded K for conv1: 243 -> 256)
    cvt_weight_k<<<256, 256, 0, stream>>>(c1w, w1b, 256, 243, 256);
    cvt_weight_k<<<20736, 256, 0, stream>>>(pw, w2b, 256, 20736, 20736);

    // conv1: M = 8*88*88 = 61952 = 968 * 64 ; relu, bf16 out
    conv_wmma_k<float, __bf16, true, 9, 9><<<968, 256, 0, stream>>>(
        x, w1b, c1b, h, 8, 3, 96, 96, 256, 88, 88, 1, 243, 256);
    // prim: M = 8*40*40 = 12800 = 200 * 64 ; K = 20736 ; fp32 out
    conv_wmma_k<__bf16, float, false, 9, 9><<<200, 256, 0, stream>>>(
        h, w2b, pb, u, 8, 256, 88, 88, 256, 40, 40, 2, 20736, 20736);

    squash_k<<<400, 256, 0, stream>>>(u);
    uhat_k<<<4000, 256, 0, stream>>>(dW, u, uh);

    zero_k<<<500, 256, 0, stream>>>(bij, 128000);
    for (int it = 0; it < 3; ++it) {
        routing1_k<<<10, 256, 0, stream>>>(bij, uh, v, out);   // out[0..79] = v
        if (it < 2) routing2_k<<<500, 256, 0, stream>>>(uh, v, bij);
    }

    dec1_k<<<16, 256, 0, stream>>>(v, w1, b1, r1);
    dec2_k<<<32, 256, 0, stream>>>(r1, w2, b2, r2);
    dec3_k<<<108, 256, 0, stream>>>(r2, w3, b3, out + 80);     // recon
}
